// GNN_85787676770949
// MI455X (gfx1250) — matrix operand order repaired and checked by reading, not compiled, not run on hardware
//
#include <hip/hip_runtime.h>

#define NNODES 100000
#define NEDGES 1600000
#define DIM    64
#define TWOD   128
#define NLAYER 3

typedef __attribute__((ext_vector_type(16))) __bf16 v16bf;
typedef __attribute__((ext_vector_type(8)))  float  v8f;

// ---------------- aggregation kernels ----------------

// aggr_ea[v][d] = selfloop attr (0..0,1); scatter_ea then atomically adds edge_attr.
__global__ void init_aggr_ea(float4* __restrict__ aggr_ea) {
  int t = blockIdx.x * blockDim.x + threadIdx.x;
  if (t >= NNODES * 16) return;
  float4 v;
  v.x = 0.f; v.y = 0.f; v.z = 0.f;
  v.w = ((t & 15) == 15) ? 1.0f : 0.f;   // last column of the row = 1
  aggr_ea[t] = v;
}

__global__ void scatter_ea(const float* __restrict__ ea,
                           const int* __restrict__ dst,
                           float* __restrict__ aggr_ea) {
  int t = blockIdx.x * blockDim.x + threadIdx.x;
  int e = t >> 4;
  if (e >= NEDGES) return;
  int c = (t & 15) * 4;
  int d = dst[e];
  const float4 v = *(const float4*)(ea + (size_t)e * DIM + c);
  float* p = aggr_ea + (size_t)d * DIM + c;
  atomicAdd(p + 0, v.x); atomicAdd(p + 1, v.y);
  atomicAdd(p + 2, v.z); atomicAdd(p + 3, v.w);
}

// self-loop contribution: aggr_h = h
__global__ void init_aggr_h(const float4* __restrict__ h, float4* __restrict__ aggr_h) {
  int t = blockIdx.x * blockDim.x + threadIdx.x;
  if (t >= NNODES * 16) return;
  aggr_h[t] = h[t];
}

__global__ void scatter_h(const float* __restrict__ h,
                          const int* __restrict__ src,
                          const int* __restrict__ dst,
                          float* __restrict__ aggr_h) {
  int t = blockIdx.x * blockDim.x + threadIdx.x;
  int e = t >> 4;
  if (e >= NEDGES) return;
  int c = (t & 15) * 4;
  int s = src[e];
  int d = dst[e];
  const float4 v = *(const float4*)(h + (size_t)s * DIM + c);  // L2-resident gather
  float* p = aggr_h + (size_t)d * DIM + c;
  atomicAdd(p + 0, v.x); atomicAdd(p + 1, v.y);
  atomicAdd(p + 2, v.z); atomicAdd(p + 3, v.w);
}

// ---------------- fused WMMA MLP ----------------
// A-frag (16x32 bf16): lane=(g,lr): a[i]   = A[lr][kk + 16g + i]     (i=0..7)
//                                   a[i+8] = A[lr][kk + 16g + 8 + i]
// B-frag (32x16 bf16): b[i] = B[kk + 16g + i][lr]   (contiguous 16 along K)
// C/D  (16x16 f32):    c[j] = C[j + 8g][lr]

__device__ __forceinline__ v16bf frag_a_f32(const float* __restrict__ p0,
                                            const float* __restrict__ p1) {
  v16bf r;
#pragma unroll
  for (int i = 0; i < 8; ++i) { r[i] = (__bf16)p0[i]; r[i + 8] = (__bf16)p1[i]; }
  return r;
}
__device__ __forceinline__ v16bf frag_a_lds(const __bf16* p0, const __bf16* p1) {
  v16bf r;
#pragma unroll
  for (int i = 0; i < 8; ++i) { r[i] = p0[i]; r[i + 8] = p1[i]; }
  return r;
}
__device__ __forceinline__ v16bf frag_b_lds(const __bf16* p) {
  v16bf r;
#pragma unroll
  for (int i = 0; i < 16; ++i) r[i] = p[i];
  return r;
}

__global__ __launch_bounds__(256)
void gnn_mlp_wmma(const float* __restrict__ aggr_ea,   // [N,64]  cols 0..63
                  const float* __restrict__ aggr_h,    // [N,64]  cols 64..127
                  const float* __restrict__ W1,        // [128,128] row=k, col=n
                  const float* __restrict__ b1,        // [128]
                  const float* __restrict__ W2,        // [128,64]
                  const float* __restrict__ b2,        // [64]
                  float* __restrict__ out,             // [N,64]
                  int relu_out) {
  __shared__ __bf16 sW1T[TWOD][TWOD];      // 32 KB : sW1T[n][k]
  __shared__ __bf16 sW2T[DIM][TWOD];       // 16 KB : sW2T[n][k]
  __shared__ float  sB1[TWOD];
  __shared__ float  sB2[DIM];
  __shared__ __bf16 sHid[8][16][TWOD];     // 32 KB : per-wave hid tile (A-operand bounce)

  const int tid = threadIdx.x;

  // stage transposed bf16 weights + biases into LDS
  for (int idx = tid; idx < TWOD * TWOD; idx += 256) {
    int k = idx >> 7, n = idx & (TWOD - 1);
    sW1T[n][k] = (__bf16)W1[idx];
  }
  for (int idx = tid; idx < TWOD * DIM; idx += 256) {
    int k = idx >> 6, n = idx & (DIM - 1);
    sW2T[n][k] = (__bf16)W2[idx];
  }
  if (tid < TWOD) sB1[tid] = b1[tid];
  if (tid < DIM)  sB2[tid] = b2[tid];
  __syncthreads();

  const int wave = tid >> 5;
  const int lane = tid & 31;
  const int g    = lane >> 4;
  const int lr   = lane & 15;

  const int m0 = blockIdx.x * 128 + wave * 16;   // 16-node tile per wave
  int mrow = m0 + lr;                            // clamp loads, predicate stores
  if (mrow >= NNODES) mrow = NNODES - 1;

  // ---- GEMM1: hid[16x128] = ReLU([aggr_ea | aggr_h] @ W1 + b1)
  v8f acc[8];
#pragma unroll
  for (int nt = 0; nt < 8; ++nt) {
    float bv = sB1[nt * 16 + lr];
#pragma unroll
    for (int j = 0; j < 8; ++j) acc[nt][j] = bv;
  }

#pragma unroll
  for (int s = 0; s < 4; ++s) {                 // K-steps of 32; steps 0,1 hit ea, 2,3 hit h
    const float* srcm = (s < 2) ? aggr_ea : aggr_h;
    const int cb = (s < 2) ? (32 * s) : (32 * s - 64);
    const float* base = srcm + (size_t)mrow * DIM + cb + 16 * g;
    v16bf a = frag_a_f32(base, base + 8);
#pragma unroll
    for (int nt = 0; nt < 8; ++nt) {
      v16bf b = frag_b_lds(&sW1T[nt * 16 + lr][32 * s + 16 * g]);
      acc[nt] = __builtin_amdgcn_wmma_f32_16x16x32_bf16(false, a, false, b,
                                                        (short)0, acc[nt], false, false);
    }
  }

  // ReLU + park hid in per-wave LDS slice (C-layout -> A-layout transpose bounce)
#pragma unroll
  for (int nt = 0; nt < 8; ++nt) {
#pragma unroll
    for (int j = 0; j < 8; ++j) {
      float v = acc[nt][j];
      v = v > 0.f ? v : 0.f;
      sHid[wave][j + 8 * g][nt * 16 + lr] = (__bf16)v;
    }
  }
  asm volatile("s_wait_dscnt 0" ::: "memory");   // same-wave LDS RAW

  // ---- GEMM2: out[16x64] = hid @ W2 + b2
  v8f acc2[4];
#pragma unroll
  for (int nt = 0; nt < 4; ++nt) {
    float bv = sB2[nt * 16 + lr];
#pragma unroll
    for (int j = 0; j < 8; ++j) acc2[nt][j] = bv;
  }
#pragma unroll
  for (int s = 0; s < 4; ++s) {
    const __bf16* ha = &sHid[wave][lr][32 * s + 16 * g];
    v16bf a = frag_a_lds(ha, ha + 8);
#pragma unroll
    for (int nt = 0; nt < 4; ++nt) {
      v16bf b = frag_b_lds(&sW2T[nt * 16 + lr][32 * s + 16 * g]);
      acc2[nt] = __builtin_amdgcn_wmma_f32_16x16x32_bf16(false, a, false, b,
                                                         (short)0, acc2[nt], false, false);
    }
  }

#pragma unroll
  for (int nt = 0; nt < 4; ++nt) {
#pragma unroll
    for (int j = 0; j < 8; ++j) {
      int m = m0 + j + 8 * g;
      if (m < NNODES) {
        float v = acc2[nt][j];
        if (relu_out) v = v > 0.f ? v : 0.f;
        out[(size_t)m * DIM + nt * 16 + lr] = v;
      }
    }
  }
}

// ---------------- launcher ----------------
extern "C" void kernel_launch(void* const* d_in, const int* in_sizes, int n_in,
                              void* d_out, int out_size, void* d_ws, size_t ws_size,
                              hipStream_t stream) {
  const float* x    = (const float*)d_in[0];
  const int*   eidx = (const int*)d_in[1];
  const float* eatt = (const float*)d_in[2];
  const float* W1   = (const float*)d_in[3];
  const float* b1   = (const float*)d_in[4];
  const float* W2   = (const float*)d_in[5];
  const float* b2   = (const float*)d_in[6];
  float* out = (float*)d_out;

  const int* srcIdx = eidx;            // edge_index[0]
  const int* dstIdx = eidx + NEDGES;   // edge_index[1]

  // workspace: aggr_ea | aggr_h | hA | hB  (4 * N*64 f32 = ~102 MB)
  float* ws      = (float*)d_ws;
  float* aggr_ea = ws;
  float* aggr_h  = ws + 1 * (size_t)NNODES * DIM;
  float* hA      = ws + 2 * (size_t)NNODES * DIM;
  float* hB      = ws + 3 * (size_t)NNODES * DIM;

  const int TPB = 256;
  dim3 gN16((NNODES * 16 + TPB - 1) / TPB);
  dim3 gE16((NEDGES * 16 + TPB - 1) / TPB);
  dim3 gMLP((NNODES + 127) / 128);

  // layer-invariant half of the segment-sum: done ONCE (saves 2 passes over 410 MB)
  init_aggr_ea<<<gN16, TPB, 0, stream>>>((float4*)aggr_ea);
  scatter_ea<<<gE16, TPB, 0, stream>>>(eatt, dstIdx, aggr_ea);

  const float* hin = x;
  for (int l = 0; l < NLAYER; ++l) {
    init_aggr_h<<<gN16, TPB, 0, stream>>>((const float4*)hin, (float4*)aggr_h);
    scatter_h<<<gE16, TPB, 0, stream>>>(hin, srcIdx, dstIdx, aggr_h);
    float* hout = (l == NLAYER - 1) ? out : ((l == 0) ? hA : hB);
    gnn_mlp_wmma<<<gMLP, TPB, 0, stream>>>(aggr_ea, aggr_h,
                                           W1 + (size_t)l * TWOD * TWOD,
                                           b1 + (size_t)l * TWOD,
                                           W2 + (size_t)l * TWOD * DIM,
                                           b2 + (size_t)l * DIM,
                                           hout, (l < NLAYER - 1) ? 1 : 0);
    hin = hout;
  }
}
